// D3_dispersion_71098888618606
// MI455X (gfx1250) — matrix operand order repaired and checked by reading, not compiled, not run on hardware
//
#include <hip/hip_runtime.h>

typedef __attribute__((ext_vector_type(16))) _Float16 v16h;
typedef __attribute__((ext_vector_type(8)))  float    v8f;

#define D3_K1      16.0f
#define D3_K3     (-4.0f)
#define D3_CUTOFF  12.0f
#define D3_CUTON   10.0f
#define D3_SMALL   1e-30f
#define D3_NREF    5
#define D3_MAXZ    95

__device__ __forceinline__ float switch_fn(float d) {
    float x = (D3_CUTOFF - d) * (1.0f / (D3_CUTOFF - D3_CUTON));
    float s = ((6.0f * x - 15.0f) * x + 10.0f) * x * x * x;
    return (d < D3_CUTON) ? 1.0f : ((d >= D3_CUTOFF) ? 0.0f : s);
}

// select c[r] with r in 0..7 (lowers to a short cndmask tree, stays in VGPRs)
__device__ __forceinline__ float sel8(const v8f c, int r) {
    float v = c[0];
    v = (r == 1) ? c[1] : v;
    v = (r == 2) ? c[2] : v;
    v = (r == 3) ? c[3] : v;
    v = (r == 4) ? c[4] : v;
    v = (r == 5) ? c[5] : v;
    v = (r == 6) ? c[6] : v;
    v = (r == 7) ? c[7] : v;
    return v;
}

// ---- phase 0: zero the accumulators (d_out and cn scratch are poisoned) ----
__global__ void d3_zero_kernel(float* __restrict__ cn, float* __restrict__ out, int n) {
    int t = blockIdx.x * blockDim.x + threadIdx.x;
    if (t < n) { cn[t] = 0.0f; out[t] = 0.0f; }
}

// ---- phase 1: per-pair CN contributions, scatter-sum to atoms ----
__global__ void d3_cn_kernel(const float* __restrict__ dist,
                             const float* __restrict__ rcov,
                             const int*   __restrict__ Z,
                             const int*   __restrict__ idx_i,
                             const int*   __restrict__ idx_j,
                             float* __restrict__ cn, int P) {
    int p = blockIdx.x * blockDim.x + threadIdx.x;
    if (p >= P) return;
    if (p + 8192 < P) {                       // stream-ahead: global_prefetch_b8
        __builtin_prefetch(&dist[p + 8192], 0, 1);
        __builtin_prefetch(&idx_i[p + 8192], 0, 1);
    }
    float d = dist[p];
    int i = idx_i[p], j = idx_j[p];
    float rcij = rcov[Z[i]] + rcov[Z[j]];
    float sw = switch_fn(d);
    float c  = sw / (1.0f + expf(-D3_K1 * (rcij / d - 1.0f)));
    atomicAdd(&cn[i], c);
}

// ---- phase 2: per-atom Gaussian weights over 5 reference CNs ----
__global__ void d3_weights_kernel(const float* __restrict__ rcn,
                                  const int*   __restrict__ Z,
                                  const float* __restrict__ cn,
                                  float* __restrict__ W, int N) {
    int n = blockIdx.x * blockDim.x + threadIdx.x;
    if (n >= N) return;
    int z = Z[n];
    float c = cn[n];
    float r[D3_NREF], g[D3_NREF];
    float rmax = -1e30f, norm = 0.0f;
#pragma unroll
    for (int k = 0; k < D3_NREF; ++k) { r[k] = rcn[z * D3_NREF + k]; rmax = fmaxf(rmax, r[k]); }
#pragma unroll
    for (int k = 0; k < D3_NREF; ++k) {
        float dc = r[k] - c;
        g[k] = (r[k] >= 0.0f) ? expf(D3_K3 * dc * dc) : 0.0f;
        norm += g[k];
    }
    float inv = 1.0f / fmaxf(norm, D3_SMALL);
#pragma unroll
    for (int k = 0; k < D3_NREF; ++k) {
        float w = (norm == 0.0f) ? ((r[k] == rmax) ? 1.0f : 0.0f) : g[k] * inv;
        W[(long)n * D3_NREF + k] = w;
    }
}

// ---- phase 3: WMMA-batched C6 + BJ-damped energy, scatter-sum to atoms ----
// 16 pairs per WMMA tile: A[m,k] = outer(w_i,w_j) of pair m (flattened 5x5,
// K padded 25->32), B[k,n] = rc6 tile of pair n. diag(A x B) = 16 C6 values
// from one v_wmma_f32_16x16x32_f16. Fragment layouts (ISA 16-bit A 16x32 /
// B 32x16), with hi = lane>>4, m = lane&15:
//   A element e -> K = 8*hi + e + (e>=8 ? 8 : 0), row M = m
//   B element e -> K = 8*hi + e + (e>=8 ? 8 : 0), col N = m
// Branching once on hi makes every K a compile-time constant.
__global__ void __launch_bounds__(256)
d3_energy_kernel(const float* __restrict__ dist,
                 const float* __restrict__ rc6,
                 const float* __restrict__ r2r4,
                 const float* __restrict__ s6p, const float* __restrict__ s8p,
                 const float* __restrict__ a1p, const float* __restrict__ a2p,
                 const int*   __restrict__ Z,
                 const int*   __restrict__ idx_i,
                 const int*   __restrict__ idx_j,
                 const float* __restrict__ W,
                 float* __restrict__ out,
                 long P, long strideWaves) {
    const int lane = threadIdx.x & 31;
    const int hi   = lane >> 4;        // which 16-lane half of the wave32
    const int m    = lane & 15;        // pair-in-tile / A row / B col
    const float s6 = s6p[0], s8 = s8p[0], a1 = a1p[0], a2 = a2p[0];
    const long gwave = (long)blockIdx.x * (blockDim.x >> 5) + (threadIdx.x >> 5);

    for (long base = gwave * 32; base < P; base += strideWaves * 32) {  // wave-uniform
#pragma unroll
        for (int tile = 0; tile < 2; ++tile) {
            long p = base + tile * 16 + m;
            bool valid = (p < P);
            long ps = valid ? p : 0;
            int i  = idx_i[ps], j = idx_j[ps];
            int zi = Z[i], zj = Z[j];
            const float* wp  = W + (long)i * D3_NREF;
            const float* wq  = W + (long)j * D3_NREF;
            const float* tbl = rc6 + ((long)zi * D3_MAXZ + zj) * (D3_NREF * D3_NREF);

            float wiv[D3_NREF], wjv[D3_NREF];
#pragma unroll
            for (int k = 0; k < D3_NREF; ++k) { wiv[k] = wp[k]; wjv[k] = wq[k]; }

            v16h a, b;
            if (hi == 0) {
#pragma unroll
                for (int e = 0; e < 16; ++e) {
                    const int kA = e + ((e >= 8) ? 8 : 0);        // 0..7, 16..23
                    a[e] = (_Float16)(wiv[kA / 5] * wjv[kA % 5]);
                    b[e] = (_Float16)tbl[kA];                     // K = 0..7, 16..23
                }
            } else {
#pragma unroll
                for (int e = 0; e < 16; ++e) {
                    const int kA = 8 + e + ((e >= 8) ? 8 : 0);    // 8..15, 24..31
                    float ua = (kA < 25) ? wiv[kA / 5] * wjv[kA % 5] : 0.0f;
                    a[e] = (_Float16)ua;
                    const int kB = 8 + e + ((e >= 8) ? 8 : 0);    // 8..15, 24..31
                    float tb = (kB < 25) ? tbl[kB] : 0.0f;
                    b[e] = (_Float16)tb;
                }
            }
            // branches rejoined: EXEC is all-1s at the matrix op
            v8f cacc = {};
            cacc = __builtin_amdgcn_wmma_f32_16x16x32_f16(
                false, a, false, b, (short)0, cacc, false, false);

            // Diagonal extraction without LDS memory or barriers:
            // D[m,m] lives in lane (m<8 ? m : m+16), register m&7.
            float dsel = sel8(cacc, lane & 7);
            int srcLane = (m < 8) ? m : (m + 16);
            float c6 = __shfl(dsel, srcLane, 32);                 // ds_bpermute_b32

            if (hi == 0 && valid) {
                float d  = dist[p];
                float sw = switch_fn(d);
                float qq = 3.0f * r2r4[zi] * r2r4[zj];
                float rr = a1 * sqrtf(qq) + a2;
                float d2 = d * d, d6 = d2 * d2 * d2, d8 = d6 * d2;
                float rr2 = rr * rr, rr6 = rr2 * rr2 * rr2, rr8 = rr6 * rr2;
                float e_pair = -0.5f * (s6 * c6 / (d6 + rr6)
                                      + s8 * qq * c6 / (d8 + rr8)) * sw;
                atomicAdd(&out[i], e_pair);
            }
        }
    }
}

extern "C" void kernel_launch(void* const* d_in, const int* in_sizes, int n_in,
                              void* d_out, int out_size, void* d_ws, size_t ws_size,
                              hipStream_t stream) {
    const float* dist = (const float*)d_in[0];
    const float* rcov = (const float*)d_in[1];
    const float* rcn  = (const float*)d_in[2];
    const float* rc6  = (const float*)d_in[3];
    const float* r2r4 = (const float*)d_in[4];
    const float* s6   = (const float*)d_in[5];
    const float* s8   = (const float*)d_in[6];
    const float* a1   = (const float*)d_in[7];
    const float* a2   = (const float*)d_in[8];
    const int*   Z    = (const int*)d_in[9];
    const int*   ii   = (const int*)d_in[10];
    const int*   jj   = (const int*)d_in[11];
    float* out = (float*)d_out;

    const int  N = in_sizes[9];
    const long P = in_sizes[0];

    float* cn = (float*)d_ws;          // N floats
    float* W  = cn + N;                // N*5 floats

    d3_zero_kernel<<<(N + 255) / 256, 256, 0, stream>>>(cn, out, N);
    d3_cn_kernel<<<(int)((P + 255) / 256), 256, 0, stream>>>(dist, rcov, Z, ii, jj, cn, (int)P);
    d3_weights_kernel<<<(N + 255) / 256, 256, 0, stream>>>(rcn, Z, cn, W, N);

    const int  blocks = 1024, threads = 256;
    const long strideWaves = (long)blocks * (threads / 32);       // 8192 waves
    d3_energy_kernel<<<blocks, threads, 0, stream>>>(
        dist, rc6, r2r4, s6, s8, a1, a2, Z, ii, jj, W, out, P, strideWaves);
}
